// mainfold_attention_71768903516340
// MI455X (gfx1250) — matrix operand order repaired and checked by reading, not compiled, not run on hardware
//
#include <hip/hip_runtime.h>
#include <math.h>

// ---------------------------------------------------------------------------
// CDNA5 (gfx1250) implementation of the manifold-attention forward pass.
// wave32; all large GEMMs via V_WMMA_F32_16X16X32_F16 with LDS-staged f16
// tiles (float4 global staging, ds_store_b64 packing, ds_load_b128 fragment
// reads); Floyd-Warshall geodesics run LDS-resident (160KB < 320KB/WGP).
// Each wave register-blocks M: 2 accumulators -> 2 v_wmma per K-step.
// ---------------------------------------------------------------------------

typedef __attribute__((ext_vector_type(16))) _Float16 v16h;
typedef __attribute__((ext_vector_type(8)))  float    v8f;

union FragH { v16h v; unsigned int u[8]; };
union FragF { v8f  v; float f[8]; };
union H4    { _Float16 h[4]; unsigned long long u; };

__device__ __forceinline__ float sigmf(float x) { return 1.0f / (1.0f + expf(-x)); }

// A fragment: 16x32 f16 tile, row-major in LDS (ld = 32 halves).
// ISA layout: lanes 0-15 -> M=lane, VGPR i<4: K=2i..2i+1, i>=4: K=2i+8..;
// lanes 16-31 add +8 to K. Packed 32-bit DS reads (k0 is even).
__device__ __forceinline__ v16h frag_a(const _Float16* s, int lane) {
  int m = lane & 15, hi = lane >> 4;
  FragH fr;
#pragma unroll
  for (int i = 0; i < 8; ++i) {
    int k0 = 2 * i + hi * 8 + ((i >= 4) ? 8 : 0);
    fr.u[i] = *reinterpret_cast<const unsigned int*>(s + m * 32 + k0);
  }
  return fr.v;
}

// B fragment: 32x16 f16 tile stored transposed in LDS as Bt[n][k] (ld = 32).
// ISA layout: lane -> N=lane&15, half j<8: K = j + 8*(lane>>4), j>=8: K = j + 8 + 8*(lane>>4).
__device__ __forceinline__ v16h frag_b(const _Float16* s, int lane, int n0) {
  int n = lane & 15, hi = lane >> 4;
  FragH fr;
#pragma unroll
  for (int i = 0; i < 8; ++i) {
    int k0 = 2 * i + hi * 8 + ((i >= 4) ? 8 : 0);
    fr.u[i] = *reinterpret_cast<const unsigned int*>(s + (n0 + n) * 32 + k0);
  }
  return fr.v;
}

__device__ __forceinline__ unsigned long long pack4(float a, float b, float c, float d) {
  H4 t;
  t.h[0] = (_Float16)a; t.h[1] = (_Float16)b;
  t.h[2] = (_Float16)c; t.h[3] = (_Float16)d;
  return t.u;
}

// ---------------------------------------------------------------------------
// Generic batched WMMA GEMM: C[b,i,j] = sum_k A[b,i,k] * B', where
//   bT=1: B' = Bm[b,j,k]   (A @ B^T)
//   bT=0: B' = Bm[b,k,j]   (A @ B)
// 224 threads = 7 waves; each wave owns one 16-column tile of a 112-column
// supertile and a 32-row M block (two accumulators, two v_wmma / K-step).
// All A/B row strides used here are multiples of 4 floats -> float4 staging.
// ---------------------------------------------------------------------------
__global__ void gemm_wmma(const float* __restrict__ A, long aBS, int aLS,
                          const float* __restrict__ Bm, long bBS, int bLS, int bT,
                          float* __restrict__ C, long cBS, int cLS,
                          int M, int N, int K)
{
  __shared__ _Float16 Al[32 * 32];
  __shared__ _Float16 Bl[112 * 32];
  int b = blockIdx.x;
  int i0 = blockIdx.y * 32;
  int j0 = blockIdx.z * 112;
  int tid = threadIdx.x, lane = tid & 31, w = tid >> 5;
  const float* Ab = A + (long)b * aBS;
  const float* Bb = Bm + (long)b * bBS;
  v8f acc0 = {}, acc1 = {};
  int ksteps = (K + 31) >> 5;
  for (int ks = 0; ks < ksteps; ++ks) {
    int kk = ks << 5;
    // ---- A tile: 32 rows x 32 k = 256 float4 ----
    for (int idx = tid; idx < 256; idx += 224) {
      int m = idx >> 3, k4 = (idx & 7) << 2;
      int gi = i0 + m, gk = kk + k4;
      unsigned long long pk;
      if (gi < M && gk + 3 < K) {
        float4 v = *reinterpret_cast<const float4*>(Ab + (long)gi * aLS + gk);
        pk = pack4(v.x, v.y, v.z, v.w);
      } else {
        float e[4];
#pragma unroll
        for (int t = 0; t < 4; ++t)
          e[t] = (gi < M && gk + t < K) ? Ab[(long)gi * aLS + gk + t] : 0.0f;
        pk = pack4(e[0], e[1], e[2], e[3]);
      }
      *reinterpret_cast<unsigned long long*>(Al + m * 32 + k4) = pk;
    }
    // ---- B tile -> transposed LDS Bt[n][k], 112 x 32 ----
    if (bT) {          // B row j contiguous in k: float4 along k -> contiguous in LDS
      for (int idx = tid; idx < 896; idx += 224) {
        int n = idx >> 3, k4 = (idx & 7) << 2;
        int gj = j0 + n, gk = kk + k4;
        unsigned long long pk;
        if (gj < N && gk + 3 < K) {
          float4 v = *reinterpret_cast<const float4*>(Bb + (long)gj * bLS + gk);
          pk = pack4(v.x, v.y, v.z, v.w);
        } else {
          float e[4];
#pragma unroll
          for (int t = 0; t < 4; ++t)
            e[t] = (gj < N && gk + t < K) ? Bb[(long)gj * bLS + gk + t] : 0.0f;
          pk = pack4(e[0], e[1], e[2], e[3]);
        }
        *reinterpret_cast<unsigned long long*>(Bl + n * 32 + k4) = pk;
      }
    } else {           // B row k contiguous in j: float4 along n -> scattered stores
      for (int idx = tid; idx < 896; idx += 224) {
        int k = idx / 28, n4 = (idx % 28) << 2;
        int gk = kk + k, gj = j0 + n4;
        if (gk < K && gj + 3 < N) {
          float4 v = *reinterpret_cast<const float4*>(Bb + (long)gk * bLS + gj);
          Bl[(n4 + 0) * 32 + k] = (_Float16)v.x;
          Bl[(n4 + 1) * 32 + k] = (_Float16)v.y;
          Bl[(n4 + 2) * 32 + k] = (_Float16)v.z;
          Bl[(n4 + 3) * 32 + k] = (_Float16)v.w;
        } else {
#pragma unroll
          for (int t = 0; t < 4; ++t) {
            float e = (gk < K && gj + t < N) ? Bb[(long)gk * bLS + gj + t] : 0.0f;
            Bl[(n4 + t) * 32 + k] = (_Float16)e;
          }
        }
      }
    }
    __syncthreads();
    v16h bf = frag_b(Bl, lane, w * 16);
    v16h a0 = frag_a(Al, lane);
    v16h a1 = frag_a(Al + 16 * 32, lane);
    acc0 = __builtin_amdgcn_wmma_f32_16x16x32_f16(false, a0, false, bf,
                                                  (short)0, acc0, false, false);
    acc1 = __builtin_amdgcn_wmma_f32_16x16x32_f16(false, a1, false, bf,
                                                  (short)0, acc1, false, false);
    __syncthreads();
  }
  int hi = lane >> 4, nl = lane & 15;
  int j = j0 + w * 16 + nl;
  if (j < N) {
    FragF f0; f0.v = acc0;
    FragF f1; f1.v = acc1;
#pragma unroll
    for (int r = 0; r < 8; ++r) {
      int i = i0 + r + 8 * hi;
      if (i < M)      C[(long)b * cBS + (long)i * cLS + j]        = f0.f[r];
      if (i + 16 < M) C[(long)b * cBS + (long)(i + 16) * cLS + j] = f1.f[r];
    }
  }
}

// ---------------------------------------------------------------------------
// Channel projection: Y[b,o,n] = inorm(relu?)(sum_c W[o,c]*X[b,c,n] + bias[o])
// K=1024, N=100 (padded 112), one workgroup = (batch b, 32 output channels),
// covers ALL pixels so instance-norm statistics are computed in-block.
// X rows are 100 floats (=25 float4, 16B-aligned row starts) -> exact float4.
// ---------------------------------------------------------------------------
__global__ void proj_wmma(const float* __restrict__ X, long xBS,
                          const float* __restrict__ W, const float* __restrict__ bias,
                          const float* __restrict__ g, const float* __restrict__ gb,
                          float* __restrict__ Y, long yBS, int doRelu)
{
  __shared__ _Float16 Al[32 * 32];
  __shared__ _Float16 Bl[112 * 32];
  __shared__ float Yl[32 * 112];
  __shared__ float muS[32], rsS[32];
  int b = blockIdx.x, o0 = blockIdx.y * 32;
  int tid = threadIdx.x, lane = tid & 31, w = tid >> 5;
  const float* Xb = X + (long)b * xBS;
  // pad columns n=100..111 are zero for the whole kernel
  for (int idx = tid; idx < 12 * 32; idx += 224) {
    int n = 100 + (idx >> 5), k = idx & 31;
    Bl[n * 32 + k] = (_Float16)0.0f;
  }
  v8f acc0 = {}, acc1 = {};
  for (int ks = 0; ks < 32; ++ks) {   // K = 1024
    int kk = ks << 5;
    for (int idx = tid; idx < 256; idx += 224) {    // W tile: 32x32 = 256 float4
      int m = idx >> 3, k4 = (idx & 7) << 2;
      float4 v = *reinterpret_cast<const float4*>(W + (long)(o0 + m) * 1024 + kk + k4);
      *reinterpret_cast<unsigned long long*>(Al + m * 32 + k4) =
          pack4(v.x, v.y, v.z, v.w);
    }
    for (int idx = tid; idx < 800; idx += 224) {    // X tile: 32 rows x 25 float4
      int k = idx / 25, n4 = (idx % 25) << 2;
      float4 v = *reinterpret_cast<const float4*>(Xb + (long)(kk + k) * 100 + n4);
      Bl[(n4 + 0) * 32 + k] = (_Float16)v.x;
      Bl[(n4 + 1) * 32 + k] = (_Float16)v.y;
      Bl[(n4 + 2) * 32 + k] = (_Float16)v.z;
      Bl[(n4 + 3) * 32 + k] = (_Float16)v.w;
    }
    __syncthreads();
    v16h bf = frag_b(Bl, lane, w * 16);
    v16h a0 = frag_a(Al, lane);
    v16h a1 = frag_a(Al + 16 * 32, lane);
    acc0 = __builtin_amdgcn_wmma_f32_16x16x32_f16(false, a0, false, bf,
                                                  (short)0, acc0, false, false);
    acc1 = __builtin_amdgcn_wmma_f32_16x16x32_f16(false, a1, false, bf,
                                                  (short)0, acc1, false, false);
    __syncthreads();
  }
  {
    FragF f0; f0.v = acc0;
    FragF f1; f1.v = acc1;
    int hi = lane >> 4, col = w * 16 + (lane & 15);
#pragma unroll
    for (int r = 0; r < 8; ++r) {
      Yl[(r + 8 * hi) * 112 + col]      = f0.f[r];
      Yl[(r + 8 * hi + 16) * 112 + col] = f1.f[r];
    }
  }
  __syncthreads();
  if (tid < 32) {
    float bb = bias[o0 + tid];
    float s = 0.0f, s2 = 0.0f;
    for (int n = 0; n < 100; ++n) { float v = Yl[tid * 112 + n] + bb; s += v; s2 += v * v; }
    float mu = s * 0.01f;
    float var = s2 * 0.01f - mu * mu;
    muS[tid] = mu;
    rsS[tid] = rsqrtf(var + 1e-5f);
  }
  __syncthreads();
  for (int idx = tid; idx < 3200; idx += 224) {
    int m = idx / 100, n = idx % 100;
    int o = o0 + m;
    float v = (Yl[m * 112 + n] + bias[o] - muS[m]) * rsS[m] * g[o] + gb[o];
    if (doRelu) v = fmaxf(v, 0.0f);
    Y[(long)b * yBS + (long)o * 100 + n] = v;
  }
}

// ---------------------------------------------------------------------------
// Isomap geodesic from a precomputed Gram matrix (WMMA-produced).
// One workgroup per batch; D (n<=200 -> 160KB) lives in LDS for the whole
// dist -> kNN(8) -> symmetrize -> Floyd-Warshall pipeline.
// mode 1: write D[:100,100:200] (100x100);  mode 2: write D[:100,100] (100).
// ---------------------------------------------------------------------------
__global__ __launch_bounds__(1024) void geodesic_kernel(
    const float* __restrict__ gram, int n,
    float* __restrict__ outM, float* __restrict__ outV, int mode)
{
  extern __shared__ float Dm[];
  const int b = blockIdx.x, tid = threadIdx.x;
  const float* Gb = gram + (long)b * n * n;
  const float INFV = 1.0e30f;
  for (int idx = tid; idx < n * n; idx += 1024) {
    int i = idx / n, j = idx % n;
    float d2 = Gb[i * n + i] + Gb[j * n + j] - 2.0f * Gb[idx];
    Dm[idx] = sqrtf(fmaxf(d2, 0.0f));
  }
  __syncthreads();
  if (tid < n) {                      // kNN (k=8), insertion top-k per row
    int i = tid;
    float bd[8]; int bj[8];
#pragma unroll
    for (int t = 0; t < 8; ++t) { bd[t] = INFV; bj[t] = -1; }
    for (int j = 0; j < n; ++j) {
      if (j == i) continue;
      float d = Dm[i * n + j];
      if (d < bd[7]) {
        int t = 7;
        while (t > 0 && bd[t - 1] > d) { bd[t] = bd[t - 1]; bj[t] = bj[t - 1]; --t; }
        bd[t] = d; bj[t] = j;
      }
    }
    for (int j = 0; j < n; ++j) Dm[i * n + j] = INFV;
#pragma unroll
    for (int t = 0; t < 8; ++t) if (bj[t] >= 0) Dm[i * n + bj[t]] = bd[t];
  }
  __syncthreads();
  if (tid < n) {                      // A = min(A, A^T); diag 0 (pair-owner i<j)
    int i = tid;
    Dm[i * n + i] = 0.0f;
    for (int j = i + 1; j < n; ++j) {
      float mn = fminf(Dm[i * n + j], Dm[j * n + i]);
      Dm[i * n + j] = mn; Dm[j * n + i] = mn;
    }
  }
  __syncthreads();
  for (int k = 0; k < n; ++k) {       // Floyd-Warshall, barrier per pivot
    for (int idx = tid; idx < n * n; idx += 1024) {
      int i = idx / n, j = idx % n;
      float alt = Dm[i * n + k] + Dm[k * n + j];
      if (alt < Dm[idx]) Dm[idx] = alt;
    }
    __syncthreads();
  }
  if (mode == 1) {
    for (int idx = tid; idx < 100 * 100; idx += 1024) {
      int i = idx / 100, j = idx % 100;
      outM[(long)b * 10000 + idx] = Dm[i * n + 100 + j];
    }
  } else {
    if (tid < 100) outV[b * 100 + tid] = Dm[tid * n + 100];
  }
}

// --------------------------- small VALU kernels ----------------------------

__global__ void joint1_kernel(const float* __restrict__ vis, const float* __restrict__ ir,
                              float* __restrict__ J)
{
  long idx = (long)blockIdx.x * blockDim.x + threadIdx.x;
  if (idx >= 13107200L) return;
  int b = (int)(idx / 204800); int r = (int)(idx % 204800);
  int i = r / 1024, c = r % 1024;
  J[idx] = (i < 100) ? vis[(long)b * 102400 + (long)c * 100 + i]
                     : ir [(long)b * 102400 + (long)c * 100 + (i - 100)];
}

__global__ void joint2_kernel(const float* __restrict__ vis, const float* __restrict__ ir,
                              const float* __restrict__ lan, const float* __restrict__ alpha,
                              const float* __restrict__ beta, float* __restrict__ J)
{
  long idx = (long)blockIdx.x * blockDim.x + threadIdx.x;
  if (idx >= 6619136L) return;
  int b = (int)(idx / 103424); int r = (int)(idx % 103424);
  int i = r / 1024, c = r % 1024;
  float v;
  if (i < 100)
    v = alpha[b] * vis[(long)b * 102400 + (long)c * 100 + i]
      + beta[b]  * ir [(long)b * 102400 + (long)c * 100 + i];
  else
    v = lan[(long)b * 1024 + c];
  J[idx] = v;
}

// M1 mix (in place) + per-batch mean -> alpha/beta gates.
__global__ void m1_alpha_kernel(float* __restrict__ M1buf,
                                const float* __restrict__ W1, const float* __restrict__ b1,
                                const float* __restrict__ lam1p,
                                const float* __restrict__ aw, const float* __restrict__ ab,
                                const float* __restrict__ bw, const float* __restrict__ bb,
                                float* __restrict__ alpha, float* __restrict__ beta)
{
  __shared__ float red[256];
  int b = blockIdx.x, tid = threadIdx.x;
  float lam = sigmf(lam1p[0]);
  float s = 0.0f;
  for (int idx = tid; idx < 10000; idx += 256) {
    float m = M1buf[(long)b * 10000 + idx];
    float v = lam * m + (1.0f - lam) * (W1[idx] * m + b1[idx]);
    M1buf[(long)b * 10000 + idx] = v;
    s += v;
  }
  red[tid] = s; __syncthreads();
  for (int st = 128; st > 0; st >>= 1) { if (tid < st) red[tid] += red[tid + st]; __syncthreads(); }
  if (tid == 0) {
    float mg = red[0] / 10000.0f;
    alpha[b] = sigmf(mg * aw[0] + ab[0]);
    beta[b]  = sigmf(mg * bw[0] + bb[0]);
  }
}

// out[b,n,c] = wa*A[b,c,n] + wb*B[b,c,n]   (transpose+gate fuse)
__global__ void fuse_kernel(const float* __restrict__ A, const float* __restrict__ Bp,
                            const float* __restrict__ alpha, const float* __restrict__ beta,
                            float* __restrict__ out, int avgMode)
{
  long idx = (long)blockIdx.x * blockDim.x + threadIdx.x;
  if (idx >= 6553600L) return;
  int b = (int)(idx / 102400); int r = (int)(idx % 102400);
  int n = r / 1024, c = r % 1024;
  float wa = avgMode ? 0.5f : alpha[b];
  float wb = avgMode ? 0.5f : beta[b];
  long src = (long)b * 102400 + (long)c * 100 + n;
  out[idx] = wa * A[src] + wb * Bp[src];
}

// Y[b,m] = act(sum_c X[b,c]*W[m,c] + bias[m])   (t_proj / t_output, GEMV)
__global__ void tproj_kernel(const float* __restrict__ X, const float* __restrict__ W,
                             const float* __restrict__ bias, float* __restrict__ Y, int doRelu)
{
  __shared__ float xin[1024];
  int b = blockIdx.x, tid = threadIdx.x;
  int m = blockIdx.y * 256 + tid;
  for (int c = tid; c < 1024; c += 256) xin[c] = X[(long)b * 1024 + c];
  __syncthreads();
  float acc = bias[m];
  const float* wr = W + (long)m * 1024;
  for (int c = 0; c < 1024; ++c) acc += xin[c] * wr[c];
  if (doRelu) acc = fmaxf(acc, 0.0f);
  Y[(long)b * 1024 + m] = acc;
}

// M_i2t[b,c] = mix(sum_n d2[b,n]*feat[b,n,c])   (feat = joint2 rows 0..99)
__global__ void mi2t_kernel(const float* __restrict__ d2, const float* __restrict__ J,
                            const float* __restrict__ W2, const float* __restrict__ b2,
                            const float* __restrict__ l2p, float* __restrict__ out)
{
  int b = blockIdx.x;
  int c = blockIdx.y * 256 + threadIdx.x;
  float acc = 0.0f;
  for (int n = 0; n < 100; ++n)
    acc += d2[(long)b * 100 + n] * J[(long)b * 103424 + (long)n * 1024 + c];
  float lam = sigmf(l2p[0]);
  out[(long)b * 1024 + c] = lam * acc + (1.0f - lam) * (W2[c] * acc + b2[c]);
}

// M_t2i[b,n,c] = mix(d2[b,n]*lan[b,c])  (rank-1 outer product + affine mix)
__global__ void mt2i_kernel(const float* __restrict__ d2, const float* __restrict__ lan,
                            const float* __restrict__ W3, const float* __restrict__ b3,
                            const float* __restrict__ l3p, float* __restrict__ out)
{
  long idx = (long)blockIdx.x * blockDim.x + threadIdx.x;
  if (idx >= 6553600L) return;
  int b = (int)(idx / 102400); int r = (int)(idx % 102400);
  int n = r / 1024, c = r % 1024;
  float fx = d2[(long)b * 100 + n] * lan[(long)b * 1024 + c];
  float lam = sigmf(l3p[0]);
  out[idx] = lam * fx + (1.0f - lam) * (W3[r] * fx + b3[r]);
}

// Aatt = (Qt@Kf^T + a1*Qt@Mt2i^T + b1*W4@G)/32 -> softmax over 100 -> Av
__global__ void aatt_kernel(const float* __restrict__ Qt, const float* __restrict__ Kf,
                            const float* __restrict__ Mt2i, const float* __restrict__ G,
                            const float* __restrict__ W4, const float* __restrict__ a1p,
                            const float* __restrict__ b1p, float* __restrict__ Av)
{
  __shared__ float red[128];
  int b = blockIdx.x, j = threadIdx.x;
  float att = -3.0e38f;
  if (j < 100) {
    const float* q  = Qt   + (long)b * 1024;
    const float* kf = Kf   + (long)b * 102400 + (long)j * 1024;
    const float* mt = Mt2i + (long)b * 102400 + (long)j * 1024;
    float a0 = 0.0f, a1 = 0.0f;
    for (int c = 0; c < 1024; ++c) { a0 += q[c] * kf[c]; a1 += q[c] * mt[c]; }
    float a2 = 0.0f;
    for (int i = 0; i < 100; ++i) a2 += W4[i] * G[(long)b * 10000 + i * 100 + j];
    att = (a0 + a1p[0] * a1 + b1p[0] * a2) * (1.0f / 32.0f);   // /sqrt(1024)
  }
  red[j] = att; __syncthreads();
  for (int st = 64; st > 0; st >>= 1) { if (j < st) red[j] = fmaxf(red[j], red[j + st]); __syncthreads(); }
  float mx = red[0]; __syncthreads();
  float e = (j < 100) ? expf(att - mx) : 0.0f;
  red[j] = e; __syncthreads();
  for (int st = 64; st > 0; st >>= 1) { if (j < st) red[j] += red[j + st]; __syncthreads(); }
  if (j < 100) Av[(long)b * 100 + j] = e / red[0];
}

// new_lan[b,m] = Av[b,:] @ Vf[b,:,m] + g1*Mi2t[b,m]*Vt[b,m]
__global__ void newlan_kernel(const float* __restrict__ Av, const float* __restrict__ Vf,
                              const float* __restrict__ Mi2t, const float* __restrict__ Vt,
                              const float* __restrict__ g1p, float* __restrict__ NL)
{
  int b = blockIdx.x, tid = threadIdx.x;
  float g1 = g1p[0];
  for (int m = tid; m < 1024; m += 256) {
    float acc = 0.0f;
    for (int j = 0; j < 100; ++j)
      acc += Av[(long)b * 100 + j] * Vf[(long)b * 102400 + (long)j * 1024 + m];
    NL[(long)b * 1024 + m] = acc + g1 * Mi2t[(long)b * 1024 + m] * Vt[(long)b * 1024 + m];
  }
}

// new_vis(pre)[b,c,n] = Vt[b,c] + g2*Mt2i[b,n,c]*Vf[b,n,c]
// (softmax over Batt's size-1 axis is identically 1 -> Bv@Vt broadcasts Vt)
__global__ void newvis_kernel(const float* __restrict__ Vt, const float* __restrict__ Mt2i,
                              const float* __restrict__ Vf, const float* __restrict__ g2p,
                              float* __restrict__ out)
{
  long idx = (long)blockIdx.x * blockDim.x + threadIdx.x;
  if (idx >= 6553600L) return;
  int b = (int)(idx / 102400); int r = (int)(idx % 102400);
  int n = r / 1024, c = r % 1024;
  float v = Vt[(long)b * 1024 + c] + g2p[0] * Mt2i[idx] * Vf[idx];
  out[(long)b * 102400 + (long)c * 100 + n] = v;
}

// ---------------------------------------------------------------------------

extern "C" void kernel_launch(void* const* d_in, const int* in_sizes, int n_in,
                              void* d_out, int out_size, void* d_ws, size_t ws_size,
                              hipStream_t stream) {
  (void)in_sizes; (void)n_in; (void)out_size; (void)ws_size;
  const float* ir   = (const float*)d_in[0];
  const float* vis  = (const float*)d_in[1];
  const float* lan  = (const float*)d_in[2];
  const float* vp_w[3] = { (const float*)d_in[3],  (const float*)d_in[7],  (const float*)d_in[11] };
  const float* vp_b[3] = { (const float*)d_in[4],  (const float*)d_in[8],  (const float*)d_in[12] };
  const float* vp_g[3] = { (const float*)d_in[5],  (const float*)d_in[9],  (const float*)d_in[13] };
  const float* vp_q[3] = { (const float*)d_in[6],  (const float*)d_in[10], (const float*)d_in[14] };
  const float* ip_w[3] = { (const float*)d_in[15], (const float*)d_in[19], (const float*)d_in[23] };
  const float* ip_b[3] = { (const float*)d_in[16], (const float*)d_in[20], (const float*)d_in[24] };
  const float* ip_g[3] = { (const float*)d_in[17], (const float*)d_in[21], (const float*)d_in[25] };
  const float* ip_q[3] = { (const float*)d_in[18], (const float*)d_in[22], (const float*)d_in[26] };
  const float* tp_w[3] = { (const float*)d_in[27], (const float*)d_in[29], (const float*)d_in[31] };
  const float* tp_b[3] = { (const float*)d_in[28], (const float*)d_in[30], (const float*)d_in[32] };
  const float* vo_w = (const float*)d_in[33];
  const float* vo_b = (const float*)d_in[34];
  const float* vo_g = (const float*)d_in[35];
  const float* vo_q = (const float*)d_in[36];
  const float* to_w = (const float*)d_in[37];
  const float* to_b = (const float*)d_in[38];
  const float* W1 = (const float*)d_in[39];
  const float* b1 = (const float*)d_in[40];
  const float* l1 = (const float*)d_in[41];
  const float* W2 = (const float*)d_in[42];
  const float* b2 = (const float*)d_in[43];
  const float* l2 = (const float*)d_in[44];
  const float* W3 = (const float*)d_in[45];
  const float* b3 = (const float*)d_in[46];
  const float* l3 = (const float*)d_in[47];
  const float* aw = (const float*)d_in[48];
  const float* ab = (const float*)d_in[49];
  const float* bw = (const float*)d_in[50];
  const float* bb = (const float*)d_in[51];
  const float* W4 = (const float*)d_in[52];
  const float* a1 = (const float*)d_in[53];
  const float* be1= (const float*)d_in[54];
  const float* g1 = (const float*)d_in[55];
  const float* g2 = (const float*)d_in[59];

  float* ws = (float*)d_ws;
  float* outVis = (float*)d_out;                 // [64,1024,10,10]
  float* outLan = (float*)d_out + 6553600L;      // [64,1,1024]

  // workspace layout (f32 element offsets, with reuse)
  float* M1    = ws + 0;          // 640,000   (M_v2i -> mixed M1 in place)
  float* Al    = ws + 640000;     // 64  alpha
  float* Be    = ws + 640064;     // 64  beta
  float* D2v   = ws + 640128;     // 6,400  (t2i == i2t, symmetric)
  float* Qt    = ws + 646528;     // 65,536
  float* Kt    = ws + 712064;     // 65,536
  float* Vt    = ws + 777600;     // 65,536
  float* Mi2t  = ws + 843136;     // 65,536
  float* Av    = ws + 908672;     // 6,400
  float* NL    = ws + 915072;     // 65,536
  float* T1    = ws + 980608;     // 640,000
  float* T2    = ws + 1620608;    // 640,000
  float* Gm    = ws + 2260608;    // 640,000  (M_t2i @ Kf^T)
  float* Gram  = ws + 2900608;    // 2,560,000 (200x200 per batch; 101x101 fits)
  float* Joint = ws + 5460608;    // 13,107,200 (joint1 / joint2 / new_vis pre)
  float* TmpA  = ws + 18567808;   // 6,553,600 (proj scratch -> M_t2i later)
  float* TmpB  = ws + 25121408;   // 6,553,600 (proj scratch -> Vf later)
  float* Qf    = ws + 31675008;   // 6,553,600
  float* Kf    = ws + 38228608;   // 6,553,600
  float* Vavg  = ws + 44782208;   // 6,553,600  (end: 51,335,808 f32 ~ 196MB)
  float* Mt2i  = TmpA;
  float* Vf    = TmpB;
  float* NVpre = Joint;

  // ---- geodesic #1 on joint1 (200 pts/batch) ----
  joint1_kernel<<<(13107200 + 255) / 256, 256, 0, stream>>>(vis, ir, Joint);
  gemm_wmma<<<dim3(64, 7, 2), 224, 0, stream>>>(Joint, 204800L, 1024,
                                                Joint, 204800L, 1024, 1,
                                                Gram, 40000L, 200, 200, 200, 1024);
  geodesic_kernel<<<64, 1024, 200 * 200 * 4, stream>>>(Gram, 200, M1, nullptr, 1);
  m1_alpha_kernel<<<64, 256, 0, stream>>>(M1, W1, b1, l1, aw, ab, bw, bb, Al, Be);

  // ---- six channel projections (WMMA + fused instance-norm) + gate fuse ----
  float* fuseDst[3] = { Qf, Kf, Vavg };
  for (int p = 0; p < 3; ++p) {
    proj_wmma<<<dim3(64, 32), 224, 0, stream>>>(vis, 102400L, vp_w[p], vp_b[p],
                                                vp_g[p], vp_q[p], TmpA, 102400L, 1);
    proj_wmma<<<dim3(64, 32), 224, 0, stream>>>(ir, 102400L, ip_w[p], ip_b[p],
                                                ip_g[p], ip_q[p], TmpB, 102400L, 1);
    fuse_kernel<<<25600, 256, 0, stream>>>(TmpA, TmpB, Al, Be, fuseDst[p], p == 2 ? 1 : 0);
  }

  // ---- text projections ----
  tproj_kernel<<<dim3(64, 4), 256, 0, stream>>>(lan, tp_w[0], tp_b[0], Qt, 1);
  tproj_kernel<<<dim3(64, 4), 256, 0, stream>>>(lan, tp_w[1], tp_b[1], Kt, 1);
  tproj_kernel<<<dim3(64, 4), 256, 0, stream>>>(lan, tp_w[2], tp_b[2], Vt, 1);

  // ---- geodesic #2 on joint2 (101 pts/batch) ----
  joint2_kernel<<<(6619136 + 255) / 256, 256, 0, stream>>>(vis, ir, lan, Al, Be, Joint);
  gemm_wmma<<<dim3(64, 4, 1), 224, 0, stream>>>(Joint, 103424L, 1024,
                                                Joint, 103424L, 1024, 1,
                                                Gram, 10201L, 101, 101, 101, 1024);
  geodesic_kernel<<<64, 1024, 101 * 101 * 4, stream>>>(Gram, 101, nullptr, D2v, 2);

  // ---- cross-modal manifold terms ----
  mi2t_kernel<<<dim3(64, 4), 256, 0, stream>>>(D2v, Joint, W2, b2, l2, Mi2t);
  mt2i_kernel<<<25600, 256, 0, stream>>>(D2v, lan, W3, b3, l3, Mt2i);

  // ---- attention chain (all WMMA) ----
  gemm_wmma<<<dim3(64, 4, 1), 224, 0, stream>>>(Qf, 102400L, 1024, Kf, 102400L, 1024, 1,
                                                T1, 10000L, 100, 100, 100, 1024);
  gemm_wmma<<<dim3(64, 4, 1), 224, 0, stream>>>(T1, 10000L, 100, M1, 10000L, 100, 0,
                                                T2, 10000L, 100, 100, 100, 100);
  gemm_wmma<<<dim3(64, 4, 10), 224, 0, stream>>>(T2, 10000L, 100, Vavg, 102400L, 1024, 0,
                                                 Vf, 102400L, 1024, 100, 1024, 100);
  gemm_wmma<<<dim3(64, 4, 1), 224, 0, stream>>>(Mt2i, 102400L, 1024, Kf, 102400L, 1024, 1,
                                                Gm, 10000L, 100, 100, 100, 1024);

  aatt_kernel<<<64, 128, 0, stream>>>(Qt, Kf, Mt2i, Gm, W4, a1, be1, Av);
  newlan_kernel<<<64, 256, 0, stream>>>(Av, Vf, Mi2t, Vt, g1, NL);
  tproj_kernel<<<dim3(64, 4), 256, 0, stream>>>(NL, to_w, to_b, outLan, 0);

  // ---- image branch output (Bv == ones -> Vt broadcast) ----
  newvis_kernel<<<25600, 256, 0, stream>>>(Vt, Mt2i, Vf, g2, NVpre);
  proj_wmma<<<dim3(64, 32), 224, 0, stream>>>(NVpre, 102400L, vo_w, vo_b,
                                              vo_g, vo_q, outVis, 102400L, 0);
}
